// DUMA_38070590112309
// MI455X (gfx1250) — matrix operand order repaired and checked by reading, not compiled, not run on hardware
//
#include <hip/hip_runtime.h>

// ---------------- types ----------------
typedef __bf16 bf16x16 __attribute__((ext_vector_type(16)));
typedef __bf16 bf16x8  __attribute__((ext_vector_type(8)));
typedef float  f32x8   __attribute__((ext_vector_type(8)));

#define SQ 512
#define HD 1024
#define NHEAD 16
#define DHEAD 64
#define MROWS 4096   // B*S

__device__ inline __bf16 f2bf(float f) {
    unsigned u = __builtin_bit_cast(unsigned, f);
    u += 0x7fffu + ((u >> 16) & 1u);
    unsigned short h = (unsigned short)(u >> 16);
    return __builtin_bit_cast(__bf16, h);
}

__device__ inline f32x8 wmma_bf16(bf16x16 a, bf16x16 b, f32x8 c) {
    return __builtin_amdgcn_wmma_f32_16x16x32_bf16(false, a, false, b, (short)0, c, false, false);
}

// A fragment (16x32 MxK): lane holds row (l%16); chunks [8*hi..+7], [16+8*hi..+7]
__device__ inline bf16x16 load_a_frag(const __bf16* base, int hi) {
    union { bf16x16 v; bf16x8 h[2]; } u;
    u.h[0] = *(const bf16x8*)(base + 8 * hi);
    u.h[1] = *(const bf16x8*)(base + 16 + 8 * hi);
    return u.v;
}

// B fragment (32x16 KxN) from [N,K]-major storage: lane holds row (l%16); chunks [8*hi..+7], [16+8*hi..+7]
__device__ inline bf16x16 load_b_frag(const __bf16* base, int hi) {
    union { bf16x16 v; bf16x8 h[2]; } u;
    u.h[0] = *(const bf16x8*)(base + 8 * hi);
    u.h[1] = *(const bf16x8*)(base + 16 + 8 * hi);
    return u.v;
}

// A fragment built from fp32 LDS row
__device__ inline bf16x16 frag_from_lds(const float* row, int k, int hi) {
    union { bf16x16 v; __bf16 e[16]; } u;
#pragma unroll
    for (int t = 0; t < 8; ++t) u.e[t] = f2bf(row[k + 8 * hi + t]);
#pragma unroll
    for (int t = 0; t < 8; ++t) u.e[8 + t] = f2bf(row[k + 16 + 8 * hi + t]);
    return u.v;
}

// ---------------- small prep kernels ----------------
__global__ void relpos_kernel(__bf16* tabP /*[144][64]*/, __bf16* tabT /*[64][160]*/) {
    for (int idx = threadIdx.x; idx < 160 * 64; idx += 256) {
        int p = idx / 64, d = idx % 64;
        float val = 0.f;
        if (p < 129) {
            int t = d >> 1;
            float freq = __expf((float)(2 * t) * (-9.210340371976184f / 64.f));
            float arg = (float)p * freq;
            val = (d & 1) ? __cosf(arg) : __sinf(arg);
        }
        if (p < 144) tabP[p * 64 + d] = f2bf(val);
        tabT[d * 160 + p] = f2bf(val);
    }
}

__global__ void cvt_kernel(const float* __restrict__ src, __bf16* __restrict__ dst, int n) {
    int i = blockIdx.x * 256 + threadIdx.x;
    if (i < n) dst[i] = f2bf(src[i]);
}

__global__ void gather_kernel(const float* __restrict__ seq, const int* __restrict__ dlen,
                              const int* __restrict__ qlen, const int* __restrict__ olen,
                              __bf16* __restrict__ out, int mode) {
    int b = blockIdx.y, j = blockIdx.x;
    int shift, valid;
    if (mode == 0) { shift = 1; valid = dlen[b]; }
    else           { shift = dlen[b] + 1; valid = qlen[b] + olen[b]; }
    int idx = j + shift; if (idx > SQ - 1) idx = SQ - 1;
    const float* src = seq + ((size_t)b * SQ + idx) * HD;
    __bf16* dst = out + ((size_t)b * SQ + j) * HD;
    bool ok = j < valid;
    for (int c = threadIdx.x; c < HD; c += 256)
        dst[c] = ok ? f2bf(src[c]) : f2bf(0.f);
}

// ---------------- WMMA GEMM: out = A[M,K](bf16) * W[N,K]^T(bf16) + bias ----------------
// flags bit0: tanh epilogue, bit1: bf16 output stored transposed per-batch [b][n][s]
__global__ __launch_bounds__(256)
void gemm_kernel(const __bf16* __restrict__ A, const __bf16* __restrict__ W,
                 const float* __restrict__ bias, const float* __restrict__ residual,
                 float* __restrict__ out_f, __bf16* __restrict__ out_b,
                 int M, int N, int K, int flags) {
    const int l = threadIdx.x & 31, wave = threadIdx.x >> 5;
    const int hi = l >> 4, ln = l & 15;
    const int wm = wave & 3, wn = wave >> 2;
    const int m0 = blockIdx.x * 128 + wm * 32;
    const int n0 = blockIdx.y * 64 + wn * 32;

    f32x8 c[2][2] = {};
    const __bf16* ar0 = A + (size_t)(m0 + ln) * K;
    const __bf16* ar1 = A + (size_t)(m0 + 16 + ln) * K;
    const __bf16* br0 = W + (size_t)(n0 + ln) * K;
    const __bf16* br1 = W + (size_t)(n0 + 16 + ln) * K;

    for (int k = 0; k < K; k += 32) {
        bf16x16 a0 = load_a_frag(ar0 + k, hi);
        bf16x16 a1 = load_a_frag(ar1 + k, hi);
        bf16x16 b0 = load_b_frag(br0 + k, hi);
        bf16x16 b1 = load_b_frag(br1 + k, hi);
        c[0][0] = wmma_bf16(a0, b0, c[0][0]);
        c[0][1] = wmma_bf16(a0, b1, c[0][1]);
        c[1][0] = wmma_bf16(a1, b0, c[1][0]);
        c[1][1] = wmma_bf16(a1, b1, c[1][1]);
    }
#pragma unroll
    for (int tm = 0; tm < 2; ++tm)
#pragma unroll
        for (int tn = 0; tn < 2; ++tn)
#pragma unroll
            for (int r = 0; r < 8; ++r) {
                int m = m0 + tm * 16 + r + 8 * hi;
                int n = n0 + tn * 16 + ln;
                float v = c[tm][tn][r] + bias[n];
                if (residual) v += residual[(size_t)m * N + n];
                if (flags & 1) v = tanhf(v);
                if (out_f) out_f[(size_t)m * N + n] = v;
                if (out_b) {
                    if (flags & 2)
                        out_b[(size_t)(m >> 9) * ((size_t)N * SQ) + (size_t)n * SQ + (m & (SQ - 1))] = f2bf(v);
                    else
                        out_b[(size_t)m * N + n] = f2bf(v);
                }
            }
}

// ---------------- fused NEZHA attention (one wave = one 16-row query tile, one head) ----------------
__global__ __launch_bounds__(32)
void attn_kernel(const __bf16* __restrict__ Q, const __bf16* __restrict__ Kmat,
                 const __bf16* __restrict__ VT /*[b][n][s]*/,
                 const __bf16* __restrict__ tabP /*[144][64]*/,
                 const __bf16* __restrict__ tabT /*[64][160]*/,
                 __bf16* __restrict__ ctx_out) {
    __shared__ float s_row[16][SQ];     // scores -> probs (fp32)
    __shared__ float p_tile[16][144];   // q . tab^T
    __shared__ float w_rel[16][160];    // rel-value bucket weights

    const int l = threadIdx.x;
    const int hi = l >> 4, ln = l & 15;
    const int it = blockIdx.x, h = blockIdx.y, b = blockIdx.z;
    const int i0 = it * 16;

    // Q fragments (K = 64 -> 2 frags)
    const __bf16* qrow = Q + ((size_t)(b * SQ + i0 + ln)) * HD + h * DHEAD;
    bf16x16 a0 = load_a_frag(qrow, hi);
    bf16x16 a1 = load_a_frag(qrow + 32, hi);

    // P tile: [16 x 144] = q_tile x tab^T
    for (int nt = 0; nt < 9; ++nt) {
        f32x8 c = {};
        const __bf16* prow = tabP + (nt * 16 + ln) * 64;
        c = wmma_bf16(a0, load_b_frag(prow, hi), c);
        c = wmma_bf16(a1, load_b_frag(prow + 32, hi), c);
#pragma unroll
        for (int r = 0; r < 8; ++r) p_tile[r + 8 * hi][nt * 16 + ln] = c[r];
    }
    __syncthreads();

    // scores: q . k^T + rel-key bias, scaled by 1/sqrt(64)
    for (int jt = 0; jt < SQ / 16; ++jt) {
        f32x8 c = {};
        const __bf16* krow = Kmat + ((size_t)(b * SQ + jt * 16 + ln)) * HD + h * DHEAD;
        c = wmma_bf16(a0, load_b_frag(krow, hi), c);
        c = wmma_bf16(a1, load_b_frag(krow + 32, hi), c);
#pragma unroll
        for (int r = 0; r < 8; ++r) {
            int il = r + 8 * hi;
            int j = jt * 16 + ln;
            int d = j - (i0 + il);
            d = (d < -64) ? -64 : (d > 64 ? 64 : d);
            s_row[il][j] = (c[r] + p_tile[il][d + 64]) * 0.125f;
        }
    }
    __syncthreads();

    // softmax over each 512-length row (wave-cooperative)
    for (int row = 0; row < 16; ++row) {
        float m = -3.4e38f;
        for (int col = l; col < SQ; col += 32) m = fmaxf(m, s_row[row][col]);
#pragma unroll
        for (int o = 16; o > 0; o >>= 1) m = fmaxf(m, __shfl_xor(m, o, 32));
        float sum = 0.f;
        for (int col = l; col < SQ; col += 32) {
            float e = __expf(s_row[row][col] - m);
            s_row[row][col] = e;
            sum += e;
        }
#pragma unroll
        for (int o = 16; o > 0; o >>= 1) sum += __shfl_xor(sum, o, 32);
        float inv = 1.f / sum;
        for (int col = l; col < SQ; col += 32) s_row[row][col] *= inv;
    }
    __syncthreads();

    // rel-value bucket weights: W[i,p] = sum_{j: clip(j-i)=p-64} probs[i,j]
    for (int row = 0; row < 16; ++row) {
        int ig = i0 + row;
        for (int p = l; p < 160; p += 32) {
            float v = 0.f;
            if (p == 0) {
                for (int j = 0; j <= ig - 64; ++j) v += s_row[row][j];
            } else if (p == 128) {
                for (int j = ig + 64; j < SQ; ++j) v += s_row[row][j];
            } else if (p < 128) {
                int j = ig + p - 64;
                if (j >= 0 && j < SQ) v = s_row[row][j];
            }
            w_rel[row][p] = v;
        }
    }
    __syncthreads();

    // ctx = probs x V  +  Wrel x tab   (accumulate in same f32 tiles)
    f32x8 cacc[4] = {};
    for (int kt = 0; kt < SQ / 32; ++kt) {
        bf16x16 ap = frag_from_lds(&s_row[ln][0], kt * 32, hi);
#pragma unroll
        for (int dt = 0; dt < 4; ++dt) {
            const __bf16* vrow = VT + ((size_t)b * HD + h * DHEAD + dt * 16 + ln) * SQ + kt * 32;
            cacc[dt] = wmma_bf16(ap, load_b_frag(vrow, hi), cacc[dt]);
        }
    }
    for (int kt = 0; kt < 5; ++kt) {
        bf16x16 ap = frag_from_lds(&w_rel[ln][0], kt * 32, hi);
#pragma unroll
        for (int dt = 0; dt < 4; ++dt) {
            const __bf16* trow = tabT + (dt * 16 + ln) * 160 + kt * 32;
            cacc[dt] = wmma_bf16(ap, load_b_frag(trow, hi), cacc[dt]);
        }
    }
#pragma unroll
    for (int dt = 0; dt < 4; ++dt)
#pragma unroll
        for (int r = 0; r < 8; ++r) {
            int il = r + 8 * hi;
            ctx_out[((size_t)(b * SQ + i0 + il)) * HD + h * DHEAD + dt * 16 + ln] = f2bf(cacc[dt][r]);
        }
}

// ---------------- LayerNorm (per 1024-wide row) ----------------
__global__ __launch_bounds__(256)
void ln_kernel(const float* __restrict__ y, const float* __restrict__ g,
               const float* __restrict__ bta, __bf16* __restrict__ out) {
    __shared__ float red[8], red2[8];
    int row = blockIdx.x;
    const float* yr = y + (size_t)row * HD;
    int wv = threadIdx.x >> 5, l = threadIdx.x & 31;

    float s = 0.f;
    for (int c = threadIdx.x; c < HD; c += 256) s += yr[c];
#pragma unroll
    for (int o = 16; o > 0; o >>= 1) s += __shfl_xor(s, o, 32);
    if (l == 0) red[wv] = s;
    __syncthreads();
    float mean = 0.f;
#pragma unroll
    for (int i = 0; i < 8; ++i) mean += red[i];
    mean *= (1.f / HD);

    float s2 = 0.f;
    for (int c = threadIdx.x; c < HD; c += 256) {
        float d = yr[c] - mean;
        s2 += d * d;
    }
#pragma unroll
    for (int o = 16; o > 0; o >>= 1) s2 += __shfl_xor(s2, o, 32);
    if (l == 0) red2[wv] = s2;
    __syncthreads();
    float var = 0.f;
#pragma unroll
    for (int i = 0; i < 8; ++i) var += red2[i];
    var *= (1.f / HD);
    float inv = rsqrtf(var + 1e-12f);

    __bf16* orow = out + (size_t)row * HD;
    for (int c = threadIdx.x; c < HD; c += 256)
        orow[c] = f2bf((yr[c] - mean) * inv * g[c] + bta[c]);
}

// ---------------- launch ----------------
extern "C" void kernel_launch(void* const* d_in, const int* in_sizes, int n_in,
                              void* d_out, int out_size, void* d_ws, size_t ws_size,
                              hipStream_t stream) {
    const float* seq  = (const float*)d_in[0];
    const int*   dlen = (const int*)d_in[1];
    const int*   qlen = (const int*)d_in[2];
    const int*   olen = (const int*)d_in[3];
    const float* Wq = (const float*)d_in[4];  const float* bq = (const float*)d_in[5];
    const float* Wk = (const float*)d_in[6];  const float* bk = (const float*)d_in[7];
    const float* Wv = (const float*)d_in[8];  const float* bv = (const float*)d_in[9];
    const float* Wp = (const float*)d_in[10]; const float* bp = (const float*)d_in[11];
    const float* Wo = (const float*)d_in[12]; const float* bo = (const float*)d_in[13];
    const float* lng = (const float*)d_in[14]; const float* lnb = (const float*)d_in[15];

    const size_t AB = (size_t)MROWS * HD * 2;   // bf16 activation buffer
    const size_t WB = (size_t)HD * HD * 2;      // bf16 weight buffer
    const size_t FB = (size_t)MROWS * HD * 4;   // fp32 activation buffer

    size_t off = 0;
    char* base = (char*)d_ws;
    auto alloc = [&](size_t bytes) -> void* {
        void* p = base + off;
        off += (bytes + 255) & ~(size_t)255;
        return p;
    };

    __bf16* tabP = (__bf16*)alloc(144 * 64 * 2);
    __bf16* tabT = (__bf16*)alloc(64 * 160 * 2);
    __bf16* wq_b = (__bf16*)alloc(WB);
    __bf16* wk_b = (__bf16*)alloc(WB);
    __bf16* wv_b = (__bf16*)alloc(WB);
    __bf16* wp_b = (__bf16*)alloc(WB);
    __bf16* wo_b = (__bf16*)alloc(WB);
    __bf16* doc_b = (__bf16*)alloc(AB);
    __bf16* qo_b  = (__bf16*)alloc(AB);
    __bf16* Qd = (__bf16*)alloc(AB);
    __bf16* Kq = (__bf16*)alloc(AB);
    __bf16* Qq = (__bf16*)alloc(AB);
    __bf16* Kd = (__bf16*)alloc(AB);
    __bf16* VT1 = (__bf16*)alloc(AB);
    __bf16* VT2 = (__bf16*)alloc(AB);
    __bf16* ctx1 = (__bf16*)alloc(AB);
    __bf16* ctx2 = (__bf16*)alloc(AB);
    __bf16* dpool = (__bf16*)alloc(AB);
    float*  qopool = (float*)alloc(FB);
    float*  y_f = (float*)alloc(FB);
    __bf16* yln = (__bf16*)alloc(AB);
    (void)ws_size; (void)in_sizes; (void)n_in; (void)out_size;

    // 1. tables + weight conversion
    relpos_kernel<<<1, 256, 0, stream>>>(tabP, tabT);
    int wn = HD * HD;
    cvt_kernel<<<(wn + 255) / 256, 256, 0, stream>>>(Wq, wq_b, wn);
    cvt_kernel<<<(wn + 255) / 256, 256, 0, stream>>>(Wk, wk_b, wn);
    cvt_kernel<<<(wn + 255) / 256, 256, 0, stream>>>(Wv, wv_b, wn);
    cvt_kernel<<<(wn + 255) / 256, 256, 0, stream>>>(Wp, wp_b, wn);
    cvt_kernel<<<(wn + 255) / 256, 256, 0, stream>>>(Wo, wo_b, wn);

    // 2. gather_shift -> bf16
    gather_kernel<<<dim3(SQ, 8), 256, 0, stream>>>(seq, dlen, qlen, olen, doc_b, 0);
    gather_kernel<<<dim3(SQ, 8), 256, 0, stream>>>(seq, dlen, qlen, olen, qo_b, 1);

    // 3. QKV projections (V stored transposed per batch)
    dim3 gg(MROWS / 128, HD / 64);
    gemm_kernel<<<gg, 256, 0, stream>>>(doc_b, wq_b, bq, nullptr, nullptr, Qd,  MROWS, HD, HD, 0);
    gemm_kernel<<<gg, 256, 0, stream>>>(qo_b,  wk_b, bk, nullptr, nullptr, Kq,  MROWS, HD, HD, 0);
    gemm_kernel<<<gg, 256, 0, stream>>>(qo_b,  wv_b, bv, nullptr, nullptr, VT1, MROWS, HD, HD, 2);
    gemm_kernel<<<gg, 256, 0, stream>>>(qo_b,  wq_b, bq, nullptr, nullptr, Qq,  MROWS, HD, HD, 0);
    gemm_kernel<<<gg, 256, 0, stream>>>(doc_b, wk_b, bk, nullptr, nullptr, Kd,  MROWS, HD, HD, 0);
    gemm_kernel<<<gg, 256, 0, stream>>>(doc_b, wv_b, bv, nullptr, nullptr, VT2, MROWS, HD, HD, 2);

    // 4. fused attention
    dim3 ag(SQ / 16, NHEAD, 8);
    attn_kernel<<<ag, 32, 0, stream>>>(Qd, Kq, VT1, tabP, tabT, ctx1);
    attn_kernel<<<ag, 32, 0, stream>>>(Qq, Kd, VT2, tabP, tabT, ctx2);

    // 5. poolers
    gemm_kernel<<<gg, 256, 0, stream>>>(ctx1, wp_b, bp, nullptr, nullptr, dpool, MROWS, HD, HD, 1);
    gemm_kernel<<<gg, 256, 0, stream>>>(ctx2, wp_b, bp, nullptr, qopool, nullptr, MROWS, HD, HD, 1);

    // 6. output dense + residual, LayerNorm, final pooler
    gemm_kernel<<<gg, 256, 0, stream>>>(dpool, wo_b, bo, qopool, y_f, nullptr, MROWS, HD, HD, 0);
    ln_kernel<<<MROWS, 256, 0, stream>>>(y_f, lng, lnb, yln);
    gemm_kernel<<<gg, 256, 0, stream>>>(yln, wp_b, bp, nullptr, (float*)d_out, nullptr, MROWS, HD, HD, 1);
}
